// SpatialGraphEncoder_50680614093460
// MI455X (gfx1250) — compile-verified, matrix operand order checked by reading, not run on hardware
//
#include <hip/hip_runtime.h>
#include <hip/hip_bf16.h>

// ---------- problem constants (match reference) ----------
#define NN      20000
#define EE      320000
#define ETOT    (EE + NN)     // edges + self loops
#define DD      512
#define HH      4
#define CC      256
#define HC      1024          // H*C
#define LL      4
#define BB      16

typedef __attribute__((ext_vector_type(16))) __bf16       bf16x16;
typedef __attribute__((ext_vector_type(8)))  float        floatx8;
typedef __attribute__((ext_vector_type(4)))  unsigned int uintx4;

union FragU { uintx4 q[2]; bf16x16 v; };

static __device__ __forceinline__ unsigned short f32_to_bf16(float f) {
    unsigned int u = __float_as_uint(f);
    u += 0x7FFFu + ((u >> 16) & 1u);   // round-to-nearest-even
    return (unsigned short)(u >> 16);
}

// Load one 16x32 bf16 fragment (A layout; B uses same layout on pre-transposed W).
// lanes 0-15:  row = r0+lane,     K halves {0..7, 16..23}
// lanes 16-31: row = r0+lane-16,  K halves {8..15, 24..31}
static __device__ __forceinline__ bf16x16 load_frag(const unsigned short* __restrict__ base,
                                                    int row, int ld, int k0, int lane) {
    const unsigned short* p = base + (size_t)row * ld + k0 + ((lane & 16) ? 8 : 0);
    FragU f;
    f.q[0] = *(const uintx4*)(p);
    f.q[1] = *(const uintx4*)(p + 16);
    return f.v;
}

// ---------- bf16 WMMA GEMM:  C[M,Ncols] = A[M,K] * B[K,Ncols]  (Bt is [Ncols,K]) ----------
// block = 256 threads = 8 waves.
// Each wave computes a 32(M) x 64(N) tile: 2 A fragments amortize 4 B fragments
// (12 b128 loads -> 8 WMMAs per k-step). blockIdx.x -> 32-row tile,
// (blockIdx.y*8 + wave) -> 64-col strip. All guards are wave-uniform (EXEC
// stays all-ones around every WMMA, per ISA restriction).
__global__ void gemm_bf16_wmma(const unsigned short* __restrict__ A,
                               const unsigned short* __restrict__ Bt,
                               float* __restrict__ Cm,
                               int M, int Ncols, int K) {
    int lane = threadIdx.x & 31;
    int wave = threadIdx.x >> 5;
    int m0 = blockIdx.x * 32;
    int n0 = (blockIdx.y * 8 + wave) * 64;
    if (m0 >= M || n0 >= Ncols) return;           // wave-uniform
    const bool haveHi = (m0 + 16) < M;            // wave-uniform

    floatx8 accLo[4] = {};
    floatx8 accHi[4] = {};
    int arow  = m0 + (lane & 15);
    int brow0 = n0 + (lane & 15);

    for (int k0 = 0; k0 < K; k0 += 32) {
        bf16x16 a0 = load_frag(A, arow, K, k0, lane);
        bf16x16 a1 = haveHi ? load_frag(A, arow + 16, K, k0, lane) : a0;
#pragma unroll
        for (int j = 0; j < 4; ++j) {
            bf16x16 b = load_frag(Bt, brow0 + j * 16, K, k0, lane);
            accLo[j] = __builtin_amdgcn_wmma_f32_16x16x32_bf16(
                false, a0, false, b, (short)0, accLo[j], false, false);
            accHi[j] = __builtin_amdgcn_wmma_f32_16x16x32_bf16(
                false, a1, false, b, (short)0, accHi[j], false, false);
        }
    }
    int nloc = lane & 15;
    int mb = m0 + ((lane & 16) ? 8 : 0);
#pragma unroll
    for (int j = 0; j < 4; ++j)
#pragma unroll
        for (int v = 0; v < 8; ++v)
            Cm[(size_t)(mb + v) * Ncols + n0 + j * 16 + nloc] = accLo[j][v];
    if (haveHi) {
#pragma unroll
        for (int j = 0; j < 4; ++j)
#pragma unroll
            for (int v = 0; v < 8; ++v)
                Cm[(size_t)(mb + 16 + v) * Ncols + n0 + j * 16 + nloc] = accHi[j][v];
    }
}

// ---------- helpers ----------
__global__ void convert_bf16_kernel(const float* __restrict__ src,
                                    unsigned short* __restrict__ dst, int total) {
    int t = blockIdx.x * blockDim.x + threadIdx.x;
    if (t < total) dst[t] = f32_to_bf16(src[t]);
}

// Wt[n][k] = bf16(W[k][n]);  W is [K, Ncols]
__global__ void transpose_w_kernel(const float* __restrict__ W,
                                   unsigned short* __restrict__ Wt, int K, int Ncols) {
    int t = blockIdx.x * blockDim.x + threadIdx.x;
    if (t >= K * Ncols) return;
    int n = t / K, k = t - n * K;
    Wt[t] = f32_to_bf16(W[(size_t)k * Ncols + n]);
}

// alpha_src/dst[n][h] = dot(h[n, h*256 ..], att[h*256 ..]); one node/block, one head/wave
__global__ void alpha_kernel(const float* __restrict__ h,
                             const float* __restrict__ a_src,
                             const float* __restrict__ a_dst,
                             float* __restrict__ aS, float* __restrict__ aD) {
    int n = blockIdx.x;
    int head = threadIdx.x >> 5;
    int lane = threadIdx.x & 31;
    const float* hp = h + (size_t)n * HC + head * CC;
    const float* as = a_src + head * CC;
    const float* ad = a_dst + head * CC;
    float s = 0.f, d = 0.f;
    for (int c = lane; c < CC; c += 32) { float hv = hp[c]; s += hv * as[c]; d += hv * ad[c]; }
    for (int off = 16; off; off >>= 1) { s += __shfl_down(s, off, 32); d += __shfl_down(d, off, 32); }
    if (lane == 0) { aS[n * HH + head] = s; aD[n * HH + head] = d; }
}

__global__ void init_softmax_kernel(unsigned int* __restrict__ m_enc,
                                    float* __restrict__ denom, int total) {
    int t = blockIdx.x * blockDim.x + threadIdx.x;
    if (t < total) { m_enc[t] = 0x007FFFFFu /* encode(-inf) */; denom[t] = 0.f; }
}

__global__ void init_out_kernel(float* __restrict__ out, const float* __restrict__ bias, int total) {
    int t = blockIdx.x * blockDim.x + threadIdx.x;
    if (t < total) out[t] = bias[t & (HC - 1)];
}

static __device__ __forceinline__ unsigned int enc_f32(float v) {
    unsigned int b = __float_as_uint(v);
    return (b & 0x80000000u) ? ~b : (b | 0x80000000u);
}
static __device__ __forceinline__ float dec_f32(unsigned int k) {
    unsigned int b = (k & 0x80000000u) ? (k & 0x7FFFFFFFu) : ~k;
    return __uint_as_float(b);
}

// pass A: e = leaky_relu(aS[src] + aD[dst]); atomic-max per (dst, head)
__global__ void edge_score_kernel(const int* __restrict__ ei,
                                  const float* __restrict__ aS, const float* __restrict__ aD,
                                  float* __restrict__ e_val, unsigned int* __restrict__ m_enc) {
    int t = blockIdx.x * blockDim.x + threadIdx.x;
    if (t >= ETOT * HH) return;
    int e = t >> 2, hh = t & 3;
    int s, d;
    if (e < EE) { s = ei[e]; d = ei[EE + e]; } else { s = d = e - EE; }
    float v = aS[s * HH + hh] + aD[d * HH + hh];
    v = v > 0.f ? v : 0.2f * v;
    e_val[t] = v;
    atomicMax(&m_enc[d * HH + hh], enc_f32(v));
}

// pass B: ex = exp(e - m[dst]); atomic-add into denom
__global__ void edge_exp_kernel(const int* __restrict__ ei,
                                float* __restrict__ e_val,
                                const unsigned int* __restrict__ m_enc,
                                float* __restrict__ denom) {
    int t = blockIdx.x * blockDim.x + threadIdx.x;
    if (t >= ETOT * HH) return;
    int e = t >> 2, hh = t & 3;
    int d = (e < EE) ? ei[EE + e] : e - EE;
    float m = dec_f32(m_enc[d * HH + hh]);
    float ex = __expf(e_val[t] - m);
    e_val[t] = ex;
    atomicAdd(&denom[d * HH + hh], ex);
}

// pass C: out[dst] += (ex/denom) * h[src]; one edge per block
__global__ void edge_aggregate_kernel(const int* __restrict__ ei,
                                      const float* __restrict__ e_val,
                                      const float* __restrict__ denom,
                                      const float* __restrict__ hg,
                                      float* __restrict__ out) {
    int e = blockIdx.x;
    int s, d;
    if (e < EE) { s = ei[e]; d = ei[EE + e]; } else { s = d = e - EE; }
    __shared__ float coef[HH];
    if (threadIdx.x < HH)
        coef[threadIdx.x] = e_val[e * HH + threadIdx.x] /
                            (denom[d * HH + threadIdx.x] + 1e-16f);
    __syncthreads();
    const float* hs = hg + (size_t)s * HC;
    float* od = out + (size_t)d * HC;
    for (int f = threadIdx.x; f < HC; f += 256)
        atomicAdd(&od[f], coef[f >> 8] * hs[f]);
}

__global__ void relu_kernel(float* __restrict__ x, int total) {
    int t = blockIdx.x * blockDim.x + threadIdx.x;
    if (t < total) x[t] = fmaxf(x[t], 0.f);
}

__global__ void init_pool_kernel(float* __restrict__ pooled, float* __restrict__ counts) {
    int t = blockIdx.x * blockDim.x + threadIdx.x;
    if (t < BB * HC) pooled[t] = 0.f;
    if (t < BB) counts[t] = 0.f;
}

__global__ void pool_kernel(const float* __restrict__ h, const int* __restrict__ batch,
                            float* __restrict__ pooled, float* __restrict__ counts) {
    int n = blockIdx.x;
    int b = batch[n];
    if (threadIdx.x == 0) atomicAdd(&counts[b], 1.0f);
    for (int f = threadIdx.x; f < HC; f += 256)
        atomicAdd(&pooled[(size_t)b * HC + f], h[(size_t)n * HC + f]);
}

__global__ void pool_to_bf16_kernel(const float* __restrict__ pooled,
                                    const float* __restrict__ counts,
                                    unsigned short* __restrict__ dst) {
    int t = blockIdx.x * blockDim.x + threadIdx.x;
    if (t >= BB * HC) return;
    int b = t >> 10;
    dst[t] = f32_to_bf16(pooled[t] / fmaxf(counts[b], 1.0f));
}

// bias + layernorm over last dim (512); one row per block, 256 threads x 2 elems
__global__ void proj_ln_kernel(const float* __restrict__ proj, const float* __restrict__ pb,
                               const float* __restrict__ g, const float* __restrict__ bet,
                               float* __restrict__ out) {
    int r = blockIdx.x;
    __shared__ float r1[256], r2[256];
    int t = threadIdx.x;
    float v0 = proj[r * DD + t] + pb[t];
    float v1 = proj[r * DD + 256 + t] + pb[256 + t];
    r1[t] = v0 + v1;
    r2[t] = v0 * v0 + v1 * v1;
    __syncthreads();
    for (int s = 128; s; s >>= 1) {
        if (t < s) { r1[t] += r1[t + s]; r2[t] += r2[t + s]; }
        __syncthreads();
    }
    float mu  = r1[0] * (1.0f / DD);
    float var = r2[0] * (1.0f / DD) - mu * mu;
    float inv = rsqrtf(var + 1e-5f);
    out[r * DD + t]       = (v0 - mu) * inv * g[t]       + bet[t];
    out[r * DD + 256 + t] = (v1 - mu) * inv * g[256 + t] + bet[256 + t];
}

// ---------- host ----------
extern "C" void kernel_launch(void* const* d_in, const int* in_sizes, int n_in,
                              void* d_out, int out_size, void* d_ws, size_t ws_size,
                              hipStream_t stream) {
    const float* x        = (const float*)d_in[0];
    const int*   ei       = (const int*)d_in[1];
    const int*   batch    = (const int*)d_in[2];
    const float* W0       = (const float*)d_in[3];
    const float* W_rest   = (const float*)d_in[4];
    const float* att_src  = (const float*)d_in[5];
    const float* att_dst  = (const float*)d_in[6];
    const float* convb    = (const float*)d_in[7];
    const float* proj_W   = (const float*)d_in[8];
    const float* proj_b   = (const float*)d_in[9];
    const float* ln_gamma = (const float*)d_in[10];
    const float* ln_beta  = (const float*)d_in[11];
    float* out = (float*)d_out;

    // workspace carve (256B aligned)
    char* ws = (char*)d_ws;
    auto carve = [&](size_t bytes) -> void* {
        void* p = ws; ws += (bytes + 255) & ~(size_t)255; return p;
    };
    float*          hA     = (float*)carve((size_t)NN * HC * 4);
    float*          hB     = (float*)carve((size_t)NN * HC * 4);
    float*          hG     = (float*)carve((size_t)NN * HC * 4);
    unsigned short* Abf    = (unsigned short*)carve((size_t)NN * HC * 2);
    unsigned short* Wt     = (unsigned short*)carve((size_t)HC * HC * 2);
    float*          aS     = (float*)carve((size_t)NN * HH * 4);
    float*          aD     = (float*)carve((size_t)NN * HH * 4);
    unsigned int*   m_enc  = (unsigned int*)carve((size_t)NN * HH * 4);
    float*          denom  = (float*)carve((size_t)NN * HH * 4);
    float*          e_val  = (float*)carve((size_t)ETOT * HH * 4);
    float*          pooled = (float*)carve((size_t)BB * HC * 4);
    float*          counts = (float*)carve((size_t)BB * 4);
    unsigned short* poolbf = (unsigned short*)carve((size_t)BB * HC * 2);
    unsigned short* pWt    = (unsigned short*)carve((size_t)DD * HC * 2);
    float*          projt  = (float*)carve((size_t)BB * DD * 4);

    const int EH = ETOT * HH;

    for (int l = 0; l < LL; ++l) {
        const float* hin  = (l == 0) ? x : ((l & 1) ? hA : hB);
        float*       hout = (l & 1) ? hB : hA;
        int          F    = (l == 0) ? DD : HC;
        const float* W    = (l == 0) ? W0 : (W_rest + (size_t)(l - 1) * HC * HC);

        // 1) convert input activations to bf16
        int totA = NN * F;
        convert_bf16_kernel<<<(totA + 255) / 256, 256, 0, stream>>>(hin, Abf, totA);
        // 2) transpose + convert W -> [HC, F] bf16
        int totW = F * HC;
        transpose_w_kernel<<<(totW + 255) / 256, 256, 0, stream>>>(W, Wt, F, HC);
        // 3) WMMA GEMM: hG[N, HC] = Abf[N, F] @ W   (32-row tiles: 20000 = 625*32)
        gemm_bf16_wmma<<<dim3(NN / 32, 2), 256, 0, stream>>>(Abf, Wt, hG, NN, HC, F);
        // 4) attention logits per node
        alpha_kernel<<<NN, 128, 0, stream>>>(hG, att_src + (size_t)l * HH * CC,
                                             att_dst + (size_t)l * HH * CC, aS, aD);
        // 5) softmax state init, output = bias
        init_softmax_kernel<<<(NN * HH + 255) / 256, 256, 0, stream>>>(m_enc, denom, NN * HH);
        init_out_kernel<<<(NN * HC + 255) / 256, 256, 0, stream>>>(hout, convb + (size_t)l * HC, NN * HC);
        // 6) per-destination softmax over edges (3 passes)
        edge_score_kernel<<<(EH + 255) / 256, 256, 0, stream>>>(ei, aS, aD, e_val, m_enc);
        edge_exp_kernel<<<(EH + 255) / 256, 256, 0, stream>>>(ei, e_val, m_enc, denom);
        edge_aggregate_kernel<<<ETOT, 256, 0, stream>>>(ei, e_val, denom, hG, hout);
        // 7) relu
        relu_kernel<<<(NN * HC + 255) / 256, 256, 0, stream>>>(hout, NN * HC);
    }

    const float* hfin = hB;  // output of layer 3

    // global mean pool
    init_pool_kernel<<<(BB * HC + 255) / 256, 256, 0, stream>>>(pooled, counts);
    pool_kernel<<<NN, 256, 0, stream>>>(hfin, batch, pooled, counts);
    pool_to_bf16_kernel<<<(BB * HC + 255) / 256, 256, 0, stream>>>(pooled, counts, poolbf);

    // final projection (WMMA, M=16 -> upper half of 32-row tile skipped uniformly) + LN
    transpose_w_kernel<<<(HC * DD + 255) / 256, 256, 0, stream>>>(proj_W, pWt, HC, DD);
    gemm_bf16_wmma<<<dim3(1, 1), 256, 0, stream>>>(poolbf, pWt, projt, BB, DD, HC);
    proj_ln_kernel<<<BB, 256, 0, stream>>>(projt, proj_b, ln_gamma, ln_beta, out);
}
